// Actor_73057393705109
// MI455X (gfx1250) — matrix operand order repaired and checked by reading, not compiled, not run on hardware
//
#include <hip/hip_runtime.h>

typedef __bf16 bf16;
typedef __attribute__((ext_vector_type(16))) __bf16 v16bf;
typedef __attribute__((ext_vector_type(8)))  __bf16 v8bf;
typedef __attribute__((ext_vector_type(8)))  float  v8f;

#define SDIM 8
#define BDIM 32
#define DDIM 5
#define TDIM 30
#define EDIM 768
#define HDIM 128
#define H4   512
#define NSEQ 160            // B*D
#define MROWS 4800          // NSEQ*TDIM
#define KCH   24            // EDIM/32

union U16 { v16bf v; v8bf h[2]; };
union V8  { v8f v; float f[8]; };

__device__ __forceinline__ float sigmoidf(float x){ return 1.0f/(1.0f+__expf(-x)); }

__device__ __forceinline__ v8f wmma_bf16(v16bf a, v16bf b, v8f c){
  return __builtin_amdgcn_wmma_f32_16x16x32_bf16(false, a, false, b, (short)0, c, false, false);
}

__device__ __forceinline__ unsigned pack2(float x, float y){
  union { bf16 h[2]; unsigned u; } u;
  u.h[0] = (bf16)x; u.h[1] = (bf16)y;
  return u.u;
}

// ---------------------------------------------------------------- k0: weights -> bf16, fragment-swizzled
// Wall/Wd layout: [s][colTile][kChunk(4)][lane][e0..15]
// Uall    layout: [s][colTile(32)][kChunk(24)][lane][e0..15]
// (lane = (j&15) + ((k&8)<<1), e = (k&7) + ((k&16)>>1))
__global__ void k0_cvt(const float* __restrict__ Wall, const float* __restrict__ Wd,
                       const float* __restrict__ Uall,
                       bf16* __restrict__ wallS, bf16* __restrict__ wdS,
                       bf16* __restrict__ uallS){
  int i = blockIdx.x*blockDim.x + threadIdx.x;
  const int n1 = SDIM*HDIM*H4;
  const int n2 = SDIM*HDIM*HDIM;
  const int n3 = SDIM*EDIM*H4;
  if (i < n1){
    int s = i / (HDIM*H4); int rem = i % (HDIM*H4);
    int k = rem / H4, j = rem % H4;
    size_t dst = ((((size_t)s*32 + (j>>4))*4 + (k>>5))*32 + ((j&15) + ((k&8)<<1)))*16 + ((k&7) + ((k&16)>>1));
    wallS[dst] = (bf16)Wall[i];
  } else if (i < n1+n2){
    int ii = i - n1;
    int s = ii / (HDIM*HDIM); int rem = ii % (HDIM*HDIM);
    int k = rem / HDIM, j = rem % HDIM;
    size_t dst = ((((size_t)s*8 + (j>>4))*4 + (k>>5))*32 + ((j&15) + ((k&8)<<1)))*16 + ((k&7) + ((k&16)>>1));
    wdS[dst] = (bf16)Wd[ii];
  } else if (i < n1+n2+n3){
    int ii = i - n1 - n2;
    int s = ii / (EDIM*H4); int rem = ii % (EDIM*H4);
    int k = rem / H4, j = rem % H4;
    size_t dst = ((((size_t)s*32 + (j>>4))*KCH + (k>>5))*32 + ((j&15) + ((k&8)<<1)))*16 + ((k&7) + ((k&16)>>1));
    uallS[dst] = (bf16)Uall[ii];
  }
}

// ---------------------------------------------------------------- k1: XU = x @ Uall + bU + ball   (bf16 WMMA)
// M ordered t-major: row r = t*NSEQ + n  (each 16-row tile has a single t).
// Block: 256 thr (8 waves), macro tile 64 rows x 512 cols (full N).
// Wave (wr = wave&1 row strip of 32, wq = wave>>1 col quarter of 128): 16 wmma/chunk.
// Output stored in k2's tile layout: XUT[s][t][task(80)][lane(32)][g(4)*8+v]
__global__ __launch_bounds__(256) void k1_xu(
    const float* __restrict__ sent, const bf16* __restrict__ uallS,
    const float* __restrict__ bU, const float* __restrict__ ball,
    float* __restrict__ XUT){
  __shared__ __align__(16) bf16 Asm[64*32];         // [row][k]
  const int s  = blockIdx.y;
  const int m0 = blockIdx.x * 64;
  const int tid = threadIdx.x;
  const int lane = tid & 31, wave = tid >> 5;
  const int wr = wave & 1, wq = wave >> 1;
  const int l15 = lane & 15;
  const int hi16 = (lane < 16) ? 0 : 1;

  v8f acc[2][8];
  v8f zero = {0.f,0.f,0.f,0.f,0.f,0.f,0.f,0.f};
  #pragma unroll
  for (int r=0;r<2;r++)
    #pragma unroll
    for (int c=0;c<8;c++) acc[r][c]=zero;

  const int acg = tid & 7;    // A stage: group of 4 cols
  const int ar  = tid >> 3;   // A stage: row 0..31 (two passes)

  for (int kc=0;kc<KCH;kc++){
    int k0 = kc*32;
    { // stage A 64x32 (f32 -> bf16), packed b64 stores
      #pragma unroll
      for (int p=0;p<2;p++){
        int row = p*32 + ar;
        int r = m0 + row;                 // t-major row index
        int n = r % NSEQ, t = r / NSEQ;
        int b = n / DDIM, d = n % DDIM;
        const float* src = sent + ((((size_t)b*SDIM + s)*DDIM + d)*TDIM + t)*EDIM + k0 + acg*4;
        if (kc + 1 < KCH) __builtin_prefetch(src + 32, 0, 1);
        float4 v = *(const float4*)src;
        uint2 w; w.x = pack2(v.x, v.y); w.y = pack2(v.z, v.w);
        *(uint2*)&Asm[row*32 + acg*4] = w;
      }
    }
    __syncthreads();
    int kb = hi16*8;
    U16 a0, a1;
    a0.h[0] = *(const v8bf*)&Asm[(wr*32 + l15)*32 + kb];
    a0.h[1] = *(const v8bf*)&Asm[(wr*32 + l15)*32 + kb + 16];
    a1.h[0] = *(const v8bf*)&Asm[(wr*32 + 16 + l15)*32 + kb];
    a1.h[1] = *(const v8bf*)&Asm[(wr*32 + 16 + l15)*32 + kb + 16];
    #pragma unroll
    for (int c=0;c<8;c++){
      int cb = wq*8 + c;
      U16 b;
      b.v = *(const v16bf*)(uallS + ((((size_t)s*32 + cb)*KCH + kc)*32 + lane)*16);
      acc[0][c]=wmma_bf16(a0.v,b.v,acc[0][c]);
      acc[1][c]=wmma_bf16(a1.v,b.v,acc[1][c]);
    }
    __syncthreads();
  }
  // store into XUT tile layout (+ fused bU + ball bias)
  #pragma unroll
  for (int rt2=0;rt2<2;rt2++){
    int r0 = m0 + wr*32 + rt2*16;
    int t = r0 / NSEQ;
    int rt = (r0 % NSEQ) >> 4;
    #pragma unroll
    for (int c=0;c<8;c++){
      int cb = wq*8 + c;
      int g = cb >> 3, c8 = cb & 7;
      int col = cb*16 + l15;
      float bias = bU[s*H4 + col] + ball[s*H4 + col];
      int task = rt*8 + c8;
      size_t base = ((((size_t)s*TDIM + t)*80 + task)*32 + lane)*32 + g*8;
      V8 rv; rv.v = acc[rt2][c];
      #pragma unroll
      for (int v=0;v<8;v++) rv.f[v] += bias;
      *(v8f*)(XUT + base) = rv.v;
    }
  }
}

// ---------------------------------------------------------------- k2: recurrent time-LSTM (per stock, WMMA)
// LDS: cL (f32) + cbf + hbf (bf16 mirrors, fragment-friendly [row][k]) = 160 KB
__global__ __launch_bounds__(1024) void k2_timelstm(
    const float* __restrict__ XUT, const bf16* __restrict__ wallS,
    const bf16* __restrict__ wdS, const float* __restrict__ bd,
    const float* __restrict__ times, float* __restrict__ outs){
  extern __shared__ float smem2[];
  float* cL  = smem2;                          // 160*128 f32
  bf16*  cbf = (bf16*)(smem2 + NSEQ*HDIM);     // 160*128 bf16
  bf16*  hbf = cbf + NSEQ*HDIM;                // 160*128 bf16
  const int s = blockIdx.x;
  const int tid = threadIdx.x;
  const int lane = tid & 31, wave = tid >> 5;
  const int l15 = lane & 15;
  const int hi16 = (lane < 16) ? 0 : 1;
  v8f zero = {0.f,0.f,0.f,0.f,0.f,0.f,0.f,0.f};

  for (int i=tid;i<NSEQ*HDIM;i+=1024){ cL[i]=0.f; cbf[i]=(bf16)0.f; hbf[i]=(bf16)0.f; }
  __syncthreads();

  for (int t=0;t<TDIM;t++){
    // ---- phase A: acc = c @ Wd   (80 tiles = 10 rowTiles x 8 colTiles)
    v8f acca[3];
    #pragma unroll
    for (int ti=0;ti<3;ti++){
      int task = wave + ti*32;
      acca[ti]=zero;
      if (task < 80){
        int rt = task >> 3, ct = task & 7;
        int arow = rt*16 + l15;
        #pragma unroll
        for (int kc=0;kc<4;kc++){
          int kb = kc*32 + hi16*8;
          U16 a;
          a.h[0] = *(const v8bf*)&cbf[arow*HDIM + kb];
          a.h[1] = *(const v8bf*)&cbf[arow*HDIM + kb + 16];
          U16 b;
          b.v = *(const v16bf*)(wdS + ((((size_t)s*8 + ct)*4 + kc)*32 + lane)*16);
          acca[ti] = wmma_bf16(a.v,b.v,acca[ti]);
        }
      }
    }
    __syncthreads();
    // ---- c_adj = c - cs1 + cs1*tt,  cs1 = tanh(acc + bd)
    #pragma unroll
    for (int ti=0;ti<3;ti++){
      int task = wave + ti*32;
      if (task < 80){
        int rt = task >> 3, ct = task & 7;
        int j = ct*16 + l15;
        float bdv = bd[s*HDIM + j];
        #pragma unroll
        for (int v=0;v<8;v++){
          int n = rt*16 + v + hi16*8;
          float cs1 = tanhf(acca[ti][v] + bdv);
          int b = n/DDIM, d = n%DDIM;
          float tt = times[(((size_t)b*SDIM + s)*DDIM + d)*TDIM + t];
          float cold = cL[n*HDIM + j];
          cL[n*HDIM + j] = cold - cs1 + cs1*tt;
        }
      }
    }
    __syncthreads();
    // ---- phase B: 4 gate tiles per task, shared A fragment from h
    v8f accg[3][4];
    #pragma unroll
    for (int ti=0;ti<3;ti++){
      int task = wave + ti*32;
      #pragma unroll
      for (int g=0;g<4;g++) accg[ti][g]=zero;
      if (task < 80){
        int rt = task >> 3, c8 = task & 7;
        int arow = rt*16 + l15;
        #pragma unroll
        for (int kc=0;kc<4;kc++){
          int kb = kc*32 + hi16*8;
          U16 a;
          a.h[0] = *(const v8bf*)&hbf[arow*HDIM + kb];
          a.h[1] = *(const v8bf*)&hbf[arow*HDIM + kb + 16];
          #pragma unroll
          for (int g=0;g<4;g++){
            int cb = g*8 + c8;
            U16 b;
            b.v = *(const v16bf*)(wallS + ((((size_t)s*32 + cb)*4 + kc)*32 + lane)*16);
            accg[ti][g]=wmma_bf16(a.v,b.v,accg[ti][g]);
          }
        }
      }
    }
    __syncthreads();
    // ---- gates + state update (gate order: f,i,o,ct); XUT holds x@U + bU + ball, tile layout
    #pragma unroll
    for (int ti=0;ti<3;ti++){
      int task = wave + ti*32;
      if (task<80){
        int rt=task>>3, c8=task&7;
        int jH = c8*16 + l15;
        size_t xbase = ((((size_t)s*TDIM + t)*80 + task)*32 + lane)*32;
        V8 xf, xi, xo, xc;
        xf.v = *(const v8f*)(XUT + xbase);
        xi.v = *(const v8f*)(XUT + xbase + 8);
        xo.v = *(const v8f*)(XUT + xbase + 16);
        xc.v = *(const v8f*)(XUT + xbase + 24);
        #pragma unroll
        for (int v=0;v<8;v++){
          int n = rt*16 + v + hi16*8;
          float f   = sigmoidf(accg[ti][0][v] + xf.f[v]);
          float ii  = sigmoidf(accg[ti][1][v] + xi.f[v]);
          float o   = sigmoidf(accg[ti][2][v] + xo.f[v]);
          float ctg = sigmoidf(accg[ti][3][v] + xc.f[v]);
          float c2 = f*cL[n*HDIM+jH] + ii*ctg;
          float h2 = o*tanhf(c2);
          cL[n*HDIM+jH] = c2;
          cbf[n*HDIM+jH] = (bf16)c2;
          hbf[n*HDIM+jH] = (bf16)h2;
          outs[(((size_t)s*NSEQ + n)*TDIM + t)*HDIM + jH] = o;
        }
      }
    }
    __syncthreads();
  }
}

// ---------------------------------------------------------------- k3: attention 1 (per stock, per sequence)
__global__ __launch_bounds__(128) void k3_attn1(
    const float* __restrict__ outs, const int* __restrict__ lens,
    const float* __restrict__ W1, const float* __restrict__ b1,
    const float* __restrict__ W2, const float* __restrict__ b2,
    const float* __restrict__ V, const float* __restrict__ bV,
    float* __restrict__ seqb){
  extern __shared__ float sm3[];
  float* W2s = sm3;                  // 128*128
  float* on  = W2s + HDIM*HDIM;      // 30*128
  float* hns = on + TDIM*HDIM;       // 128
  float* red = hns + HDIM;           // 128
  float* sc  = red + HDIM;           // 32
  int s = blockIdx.x / NSEQ;
  int n = blockIdx.x % NSEQ;
  int j = threadIdx.x;
  for (int i=j;i<HDIM*HDIM;i+=HDIM) W2s[i]=W2[(size_t)s*HDIM*HDIM + i];
  for (int i=j;i<TDIM*HDIM;i+=HDIM) on[i]=outs[(((size_t)s*NSEQ+n)*TDIM)*HDIM + i];
  int b=n/DDIM, d=n%DDIM;
  int len = lens[(b*SDIM+s)*DDIM+d];
  __syncthreads();
  int idx = len-1; if(idx<0)idx=0; if(idx>TDIM-1)idx=TDIM-1;
  hns[j] = (len>0)? on[idx*HDIM+j] : 0.f;
  __syncthreads();
  float s1 = b1[s*HDIM+j];
  for(int k=0;k<HDIM;k++) s1 += hns[k]*W1[((size_t)s*HDIM+k)*HDIM+j];
  float vj  = V[s*HDIM + j];
  float b2j = b2[s*HDIM + j];
  for (int t=0;t<TDIM;t++){
    float acc = s1 + b2j;
    for(int k=0;k<HDIM;k++) acc += on[t*HDIM+k]*W2s[k*HDIM+j];
    red[j] = tanhf(acc)*vj;
    __syncthreads();
    for(int off=64;off>0;off>>=1){ if(j<off) red[j]+=red[j+off]; __syncthreads(); }
    if(j==0) sc[t] = red[0] + bV[s];
    __syncthreads();
  }
  float mx=-1e30f;
  for(int t=0;t<TDIM;t++) if(t<len && sc[t]>mx) mx=sc[t];
  float denom=0.f, cj=0.f;
  for(int t=0;t<TDIM;t++) if(t<len){ float w=__expf(sc[t]-mx); denom+=w; cj+=w*on[t*HDIM+j]; }
  seqb[((size_t)s*NSEQ+n)*HDIM+j] = cj/denom;
}

// ---------------------------------------------------------------- k4: LSTM2 + attention2 + dense head (per stock)
__global__ __launch_bounds__(256) void k4_lstm2(
    const float* __restrict__ seqb,
    const float* __restrict__ Wih, const float* __restrict__ bih,
    const float* __restrict__ Whh, const float* __restrict__ bhh,
    const float* __restrict__ W1, const float* __restrict__ b1,
    const float* __restrict__ W2, const float* __restrict__ b2,
    const float* __restrict__ V, const float* __restrict__ bV,
    const float* __restrict__ x1W, const float* __restrict__ x1b,
    const float* __restrict__ x2W, const float* __restrict__ x2b,
    float* __restrict__ text){
  extern __shared__ float sm4[];
  float* seqs = sm4;                      // 32*5*128 = 20480
  float* hss  = seqs + 20480;             // 20480
  float* hLl  = hss + 20480;              // 4096
  float* cLl  = hLl + 4096;               // 4096
  float* s1a  = cLl + 4096;               // 4096
  float* ctxL = s1a + 4096;               // 4096
  float* yL   = ctxL + 4096;              // 4096
  float* sc2  = yL + 4096;                // 160
  int s = blockIdx.x;
  int tid = threadIdx.x;
  for(int i=tid;i<BDIM*DDIM*HDIM;i+=256) seqs[i]=seqb[(size_t)s*BDIM*DDIM*HDIM + i];
  for(int i=tid;i<BDIM*HDIM;i+=256){ hLl[i]=0.f; cLl[i]=0.f; }
  __syncthreads();
  // LSTM2 (gate order: i,f,g,o)
  for(int d=0; d<DDIM; d++){
    for(int p=0;p<16;p++){
      int idx = tid + p*256;
      int n = idx >> 7, jH = idx & 127;
      float gi = bih[s*H4 + jH]       + bhh[s*H4 + jH];
      float gf = bih[s*H4 + 128 + jH] + bhh[s*H4 + 128 + jH];
      float gg = bih[s*H4 + 256 + jH] + bhh[s*H4 + 256 + jH];
      float go = bih[s*H4 + 384 + jH] + bhh[s*H4 + 384 + jH];
      for(int k=0;k<HDIM;k++){
        float xk = seqs[(n*DDIM + d)*HDIM + k];
        float hk = hLl[n*HDIM + k];
        const float* wih = Wih + ((size_t)s*HDIM + k)*H4;
        const float* whh = Whh + ((size_t)s*HDIM + k)*H4;
        gi += xk*wih[jH]     + hk*whh[jH];
        gf += xk*wih[128+jH] + hk*whh[128+jH];
        gg += xk*wih[256+jH] + hk*whh[256+jH];
        go += xk*wih[384+jH] + hk*whh[384+jH];
      }
      float c2 = sigmoidf(gf)*cLl[n*HDIM+jH] + sigmoidf(gi)*tanhf(gg);
      cLl[n*HDIM+jH] = c2;
      hss[(n*DDIM + d)*HDIM + jH] = sigmoidf(go)*tanhf(c2);
    }
    __syncthreads();
    for(int p=0;p<16;p++){
      int idx = tid + p*256; int n=idx>>7, jH=idx&127;
      hLl[n*HDIM+jH] = hss[(n*DDIM+d)*HDIM+jH];
    }
    __syncthreads();
  }
  // attention 2: s1a = hlast@W1 + b1
  for(int p=0;p<16;p++){
    int idx=tid+p*256; int n=idx>>7, jH=idx&127;
    float acc = b1[s*HDIM+jH];
    for(int k=0;k<HDIM;k++) acc += hLl[n*HDIM+k]*W1[((size_t)s*HDIM+k)*HDIM+jH];
    s1a[idx] = acc;
  }
  __syncthreads();
  if (tid < BDIM*DDIM){
    int n = tid/DDIM, t = tid%DDIM;
    float scv = bV[s];
    for(int jH=0;jH<HDIM;jH++){
      float a = s1a[n*HDIM+jH] + b2[s*HDIM+jH];
      for(int k=0;k<HDIM;k++) a += hss[(n*DDIM+t)*HDIM+k]*W2[((size_t)s*HDIM+k)*HDIM+jH];
      scv += tanhf(a)*V[s*HDIM+jH];
    }
    sc2[tid] = scv;
  }
  __syncthreads();
  for(int p=0;p<16;p++){
    int idx=tid+p*256; int n=idx>>7, jH=idx&127;
    float mx=-1e30f;
    for(int t=0;t<DDIM;t++) mx = fmaxf(mx, sc2[n*DDIM+t]);
    float denom=0.f, cj=0.f;
    for(int t=0;t<DDIM;t++){ float w=__expf(sc2[n*DDIM+t]-mx); denom+=w; cj+=w*hss[(n*DDIM+t)*HDIM+jH]; }
    ctxL[idx] = cj/denom;
  }
  __syncthreads();
  // y = relu(ctx2 @ x1W + x1b)
  for(int p=0;p<16;p++){
    int idx=tid+p*256; int n=idx>>7, jH=idx&127;
    float acc = x1b[s*HDIM+jH];
    for(int k=0;k<HDIM;k++) acc += ctxL[n*HDIM+k]*x1W[((size_t)s*HDIM+k)*HDIM+jH];
    yL[idx] = fmaxf(acc, 0.f);
  }
  __syncthreads();
  // per = y @ x2W + x2b  ->  text[b][s*64+jo]
  for(int p=0;p<8;p++){
    int idx=tid+p*256;          // 0..2047 = 32*64
    int n = idx>>6, jo = idx&63;
    float acc = x2b[s*64+jo];
    for(int k=0;k<HDIM;k++) acc += yL[n*HDIM+k]*x2W[((size_t)s*HDIM+k)*64+jo];
    text[(size_t)n*(SDIM*64) + s*64 + jo] = acc;
  }
}

// ---------------------------------------------------------------- k5: stock MLP + concat + tanh head
__global__ __launch_bounds__(256) void k5_final(
    const float* __restrict__ sf, const float* __restrict__ text,
    const float* __restrict__ h1W, const float* __restrict__ h1b,
    const float* __restrict__ h2W, const float* __restrict__ h2b,
    const float* __restrict__ hcW, const float* __restrict__ hcb,
    float* __restrict__ out){
  __shared__ float sfs[BDIM*17];
  __shared__ float r1[BDIM*64];
  __shared__ float xs[BDIM*32];
  int tid=threadIdx.x;
  for(int i=tid;i<BDIM*17;i+=256) sfs[i]=sf[i];
  __syncthreads();
  for(int p=0;p<8;p++){
    int idx=tid+p*256; int b=idx>>6, j=idx&63;
    float acc=h1b[j];
    for(int k=0;k<17;k++) acc += sfs[b*17+k]*h1W[k*64+j];
    r1[idx]=fmaxf(acc,0.f);
  }
  __syncthreads();
  for(int p=0;p<4;p++){
    int idx=tid+p*256; int b=idx>>5, j=idx&31;
    float acc=h2b[j];
    for(int k=0;k<64;k++) acc += r1[b*64+k]*h2W[k*32+j];
    xs[idx]=acc;
  }
  __syncthreads();
  {
    int b=tid>>3, k=tid&7;
    float acc=hcb[k];
    for(int q=0;q<32;q++)  acc += xs[b*32+q]*hcW[q*8+k];
    for(int q=0;q<512;q++) acc += text[b*512+q]*hcW[(32+q)*8+k];
    out[tid]=tanhf(acc);
  }
}

// ---------------------------------------------------------------- launcher
extern "C" void kernel_launch(void* const* d_in, const int* in_sizes, int n_in,
                              void* d_out, int out_size, void* d_ws, size_t ws_size,
                              hipStream_t stream){
  (void)in_sizes; (void)n_in; (void)out_size; (void)ws_size;
  const float* stock_feats = (const float*)d_in[0];
  const float* sent        = (const float*)d_in[1];
  const float* times       = (const float*)d_in[2];
  const int*   lens        = (const int*)d_in[3];
  const float* tl_Wall     = (const float*)d_in[4];
  const float* tl_ball     = (const float*)d_in[5];
  const float* tl_Uall     = (const float*)d_in[6];
  const float* tl_bU       = (const float*)d_in[7];
  const float* tl_Wd       = (const float*)d_in[8];
  const float* tl_bd       = (const float*)d_in[9];
  const float* a1_W1       = (const float*)d_in[10];
  const float* a1_b1       = (const float*)d_in[11];
  const float* a1_W2       = (const float*)d_in[12];
  const float* a1_b2       = (const float*)d_in[13];
  const float* a1_V        = (const float*)d_in[14];
  const float* a1_bV       = (const float*)d_in[15];
  const float* l2_Wih      = (const float*)d_in[16];
  const float* l2_bih      = (const float*)d_in[17];
  const float* l2_Whh      = (const float*)d_in[18];
  const float* l2_bhh      = (const float*)d_in[19];
  const float* a2_W1       = (const float*)d_in[20];
  const float* a2_b1       = (const float*)d_in[21];
  const float* a2_W2       = (const float*)d_in[22];
  const float* a2_b2       = (const float*)d_in[23];
  const float* a2_V        = (const float*)d_in[24];
  const float* a2_bV       = (const float*)d_in[25];
  const float* x1_W        = (const float*)d_in[26];
  const float* x1_b        = (const float*)d_in[27];
  const float* x2_W        = (const float*)d_in[28];
  const float* x2_b        = (const float*)d_in[29];
  const float* h1_W        = (const float*)d_in[30];
  const float* h1_b        = (const float*)d_in[31];
  const float* h2_W        = (const float*)d_in[32];
  const float* h2_b        = (const float*)d_in[33];
  const float* hc_W        = (const float*)d_in[34];
  const float* hc_b        = (const float*)d_in[35];

  float* wsf = (float*)d_ws;
  size_t oXU   = 0;
  size_t oOuts = oXU   + (size_t)SDIM*MROWS*H4;          // 19,660,800 f
  size_t oSeq  = oOuts + (size_t)SDIM*NSEQ*TDIM*HDIM;    // +4,915,200 f
  size_t oText = oSeq  + (size_t)SDIM*NSEQ*HDIM;         // +163,840 f
  size_t fend  = oText + (size_t)BDIM*SDIM*64;           // +16,384 f
  float* XUT  = wsf + oXU;
  float* outs = wsf + oOuts;
  float* seqb = wsf + oSeq;
  float* text = wsf + oText;
  // bf16 region, 64B aligned
  size_t fbytes = ((fend*sizeof(float)) + 63) & ~(size_t)63;
  bf16* wallS = (bf16*)((char*)d_ws + fbytes);
  bf16* wdS   = wallS + (size_t)SDIM*HDIM*H4;
  bf16* uallS = wdS   + (size_t)SDIM*HDIM*HDIM;

  int cvtTot = SDIM*HDIM*H4 + SDIM*HDIM*HDIM + SDIM*EDIM*H4;
  k0_cvt<<<(cvtTot + 255)/256, 256, 0, stream>>>(tl_Wall, tl_Wd, tl_Uall, wallS, wdS, uallS);

  k1_xu<<<dim3(75, SDIM), 256, 0, stream>>>(sent, uallS, tl_bU, tl_ball, XUT);

  size_t sm2 = (size_t)NSEQ*HDIM*(sizeof(float) + 2*sizeof(bf16));   // 160 KB
  k2_timelstm<<<SDIM, 1024, sm2, stream>>>(XUT, wallS, wdS, tl_bd, times, outs);

  size_t sm3 = (size_t)(HDIM*HDIM + TDIM*HDIM + HDIM + HDIM + 32)*sizeof(float);
  k3_attn1<<<SDIM*NSEQ, HDIM, sm3, stream>>>(outs, lens, a1_W1, a1_b1, a1_W2, a1_b2, a1_V, a1_bV, seqb);

  size_t sm4 = (size_t)(20480*2 + 4096*5 + 160)*sizeof(float);
  k4_lstm2<<<SDIM, 256, sm4, stream>>>(seqb, l2_Wih, l2_bih, l2_Whh, l2_bhh,
                                       a2_W1, a2_b1, a2_W2, a2_b2, a2_V, a2_bV,
                                       x1_W, x1_b, x2_W, x2_b, text);

  k5_final<<<1, 256, 0, stream>>>(stock_feats, text, h1_W, h1_b, h2_W, h2_b, hc_W, hc_b, (float*)d_out);
}
